// SubEncoder_1924145348933
// MI455X (gfx1250) — matrix operand order repaired and checked by reading, not compiled, not run on hardware
//
#include <hip/hip_runtime.h>
#include <hip/hip_bf16.h>
#include <stdint.h>

// ---------------------------------------------------------------------------
// Problem constants (from the reference: B=4, L=1024, D=1024, H=16, Dff=4096)
// ---------------------------------------------------------------------------
#define B_    4
#define L_    1024
#define D_    1024
#define H_    16
#define DH_   64
#define DFF_  4096
#define NT_   (B_ * L_)          // 4096 tokens

typedef _Float16 half_t;
typedef __attribute__((ext_vector_type(16))) _Float16 v16h;
typedef __attribute__((ext_vector_type(8)))  _Float16 h8;
typedef __attribute__((ext_vector_type(8)))  float    v8f;
typedef __attribute__((ext_vector_type(4)))  uint32_t v4u;
typedef __attribute__((ext_vector_type(8)))  int      v8i;
typedef __attribute__((ext_vector_type(4)))  int      v4i;

#if defined(__has_builtin)
#  if __has_builtin(__builtin_amdgcn_tensor_load_to_lds)
#    define HAVE_TDM 1
#  else
#    define HAVE_TDM 0
#  endif
#else
#  define HAVE_TDM 0
#endif

__device__ __forceinline__ v16h mk16(h8 lo, h8 hi) {
  v16h r;
#pragma unroll
  for (int i = 0; i < 8; ++i) { r[i] = lo[i]; r[i + 8] = hi[i]; }
  return r;
}

__device__ __forceinline__ v8f vzero8() {
  v8f z;
#pragma unroll
  for (int i = 0; i < 8; ++i) z[i] = 0.0f;
  return z;
}

#if HAVE_TDM
// ---------------------------------------------------------------------------
// Tensor Data Mover: 2D tile (tile_d0 x tile_d1 elements, 2-byte data) from a
// row-major tensor (row length stride_d0 elements) into LDS at lds_byte_addr.
// D# layout per CDNA5 ISA (group0 128b, group1 256b).
// This toolchain exposes the 6-arg builtin:
//   (uint32x4 g0, int32x8 g1, int32x4 g2, int32x4 g3, int32x8 g4, i32 cpol)
// ---------------------------------------------------------------------------
__device__ __forceinline__ void tdm_load_tile_2d(uint32_t lds_byte_addr,
                                                 const half_t* gptr,
                                                 uint32_t tensor_d0,
                                                 uint32_t tensor_d1,
                                                 uint32_t stride_d0,
                                                 uint32_t tile_d0,
                                                 uint32_t tile_d1) {
  const uint64_t ga = (uint64_t)(uintptr_t)gptr;
  v4u g0;
  g0[0] = 1u;                                                   // count=1 (valid user D#)
  g0[1] = lds_byte_addr;                                        // lds_addr
  g0[2] = (uint32_t)ga;                                         // global_addr[31:0]
  g0[3] = ((uint32_t)(ga >> 32) & 0x01FFFFFFu) | (2u << 30);    // addr[56:32], type=2
  v8i g1;
  g1[0] = (int)(1u << 16);                                      // data_size=1 (2 bytes)
  g1[1] = (int)((tensor_d0 & 0xFFFFu) << 16);                   // tensor_dim0[15:0]
  g1[2] = (int)((tensor_d0 >> 16) | ((tensor_d1 & 0xFFFFu) << 16));
  g1[3] = (int)((tensor_d1 >> 16) | (tile_d0 << 16));           // tile_dim0
  g1[4] = (int)(tile_d1 & 0xFFFFu);                             // tile_dim1 (tile_dim2=0)
  g1[5] = (int)stride_d0;                                       // tensor_dim0_stride[31:0]
  g1[6] = 0;
  g1[7] = 0;
  const v4i gz4 = {0, 0, 0, 0};                                 // 2D: groups 2/3 zero
  const v8i gz8 = {0, 0, 0, 0, 0, 0, 0, 0};
  __builtin_amdgcn_tensor_load_to_lds(g0, g1, gz4, gz4, gz8, 0);
}
#endif

// ---------------------------------------------------------------------------
// fp32 -> f16 cast
// ---------------------------------------------------------------------------
__global__ void cast_f16_kernel(const float* __restrict__ src,
                                half_t* __restrict__ dst, int n) {
  int i = blockIdx.x * blockDim.x + threadIdx.x;
  if (i < n) dst[i] = (half_t)src[i];
}

// ---------------------------------------------------------------------------
// fp32 [K][N] -> f16 [N][K] cast + transpose (weights only; one-time cost).
// 32x32 tiles through LDS, 256 threads (32 cols x 8 rows).
// ---------------------------------------------------------------------------
__launch_bounds__(256)
__global__ void cast_transpose_f16_kernel(const float* __restrict__ src,
                                          half_t* __restrict__ dst,
                                          int K, int N) {
  __shared__ float tile[32][33];
  const int k0 = blockIdx.y * 32;
  const int n0 = blockIdx.x * 32;
  const int tx = threadIdx.x & 31;
  const int ty = threadIdx.x >> 5;       // 0..7
#pragma unroll
  for (int i = ty; i < 32; i += 8)
    tile[i][tx] = src[(size_t)(k0 + i) * N + n0 + tx];
  __syncthreads();
#pragma unroll
  for (int i = ty; i < 32; i += 8)
    dst[(size_t)(n0 + i) * K + k0 + tx] = (half_t)tile[tx][i];
}

// ---------------------------------------------------------------------------
// Generic f16 WMMA GEMM: C[M,N] = A[M,K] @ Bt[N,K]^T (+bias, relu)
// 256 threads = 8 waves; block tile 64x64; K-step 32.
// A tile and Bt tile are both row-major 64x32 rectangles -> staged into LDS
// by the Tensor Data Mover (double-buffered, TENSORcnt-tracked) when the
// builtin exists, else by contiguous b128 global->LDS copies.
// Wave w: n-tile = w&3, m-tiles = (w>>2)*2 + {0,1}.
// ---------------------------------------------------------------------------
__launch_bounds__(256)
__global__ void gemm_f16_wmma(const half_t* __restrict__ A,
                              const half_t* __restrict__ Bt,
                              const float* __restrict__ bias,
                              float* __restrict__ Cf,
                              half_t* __restrict__ Ch,
                              int M, int N, int K, int relu) {
  const int tid    = threadIdx.x;
  const int lane   = tid & 31;
  const int wave   = tid >> 5;          // 0..7
  const int m0     = blockIdx.y * 64;
  const int n0     = blockIdx.x * 64;
  const int wn     = wave & 3;          // n-tile index
  const int wmBase = (wave >> 2) * 2;   // first m-tile index
  const int rsel   = lane & 15;
  const int hiHalf = (lane >= 16);
  const int koff   = hiHalf ? 8 : 0;

  v8f acc0 = vzero8();
  v8f acc1 = vzero8();

#if HAVE_TDM
  __shared__ __align__(16) half_t As[2][64 * 32];
  __shared__ __align__(16) half_t Bts[2][64 * 32];

  const uint32_t ldsA[2] = {(uint32_t)(uintptr_t)&As[0][0],
                            (uint32_t)(uintptr_t)&As[1][0]};
  const uint32_t ldsB[2] = {(uint32_t)(uintptr_t)&Bts[0][0],
                            (uint32_t)(uintptr_t)&Bts[1][0]};

  if (wave == 0) {    // prologue: fill buffer 0
    tdm_load_tile_2d(ldsA[0], A  + (size_t)m0 * K, (uint32_t)K, (uint32_t)M,
                     (uint32_t)K, 32u, 64u);
    tdm_load_tile_2d(ldsB[0], Bt + (size_t)n0 * K, (uint32_t)K, (uint32_t)N,
                     (uint32_t)K, 32u, 64u);
  }

  int buf = 0;
  for (int k0 = 0; k0 < K; k0 += 32, buf ^= 1) {
    if (wave == 0) {
      if (k0 + 32 < K) {            // kick off next tile pair into other buffer
        const int nb = buf ^ 1;
        tdm_load_tile_2d(ldsA[nb], A  + (size_t)m0 * K + (k0 + 32),
                         (uint32_t)K, (uint32_t)M, (uint32_t)K, 32u, 64u);
        tdm_load_tile_2d(ldsB[nb], Bt + (size_t)n0 * K + (k0 + 32),
                         (uint32_t)K, (uint32_t)N, (uint32_t)K, 32u, 64u);
        __builtin_amdgcn_s_wait_tensorcnt(2);   // current pair complete
      } else {
        __builtin_amdgcn_s_wait_tensorcnt(0);   // last tile: drain
      }
    }
    __syncthreads();                            // tile visible to all waves

    const half_t* Ab = As[buf];
    const half_t* Bb = Bts[buf];
    {
      const h8* p = (const h8*)(Bb + (wn * 16 + rsel) * 32 + koff);
      v16h bfrag = mk16(p[0], p[2]);
      const half_t* r0 = Ab + ((wmBase + 0) * 16 + rsel) * 32;
      const half_t* r1 = Ab + ((wmBase + 1) * 16 + rsel) * 32;
      v16h a0 = mk16(*(const h8*)(r0 + koff), *(const h8*)(r0 + 16 + koff));
      v16h a1 = mk16(*(const h8*)(r1 + koff), *(const h8*)(r1 + 16 + koff));
      acc0 = __builtin_amdgcn_wmma_f32_16x16x32_f16(false, a0, false, bfrag,
                                                    (short)0, acc0, false, false);
      acc1 = __builtin_amdgcn_wmma_f32_16x16x32_f16(false, a1, false, bfrag,
                                                    (short)0, acc1, false, false);
    }
    __syncthreads();                // compute done before buffer reuse
  }
#else
  __shared__ __align__(16) half_t As1[64 * 32];
  __shared__ __align__(16) half_t Bts1[64 * 32];

  const int cr = tid >> 2;          // 0..63 (tile row)
  const int cc = (tid & 3) * 8;     // 0,8,16,24

  for (int k0 = 0; k0 < K; k0 += 32) {
    h8 av = *(const h8*)(A  + (size_t)(m0 + cr) * K + k0 + cc);
    h8 bv = *(const h8*)(Bt + (size_t)(n0 + cr) * K + k0 + cc);
    __syncthreads();
    *(h8*)(&As1[cr * 32 + cc])  = av;
    *(h8*)(&Bts1[cr * 32 + cc]) = bv;
    __syncthreads();
    if (k0 + 32 < K) {
      __builtin_prefetch((const void*)(A  + (size_t)(m0 + cr) * K + k0 + 32 + cc), 0, 0);
      __builtin_prefetch((const void*)(Bt + (size_t)(n0 + cr) * K + k0 + 32 + cc), 0, 0);
    }
    {
      const h8* p = (const h8*)(Bts1 + (wn * 16 + rsel) * 32 + koff);
      v16h bfrag = mk16(p[0], p[2]);
      const half_t* r0 = As1 + ((wmBase + 0) * 16 + rsel) * 32;
      const half_t* r1 = As1 + ((wmBase + 1) * 16 + rsel) * 32;
      v16h a0 = mk16(*(const h8*)(r0 + koff), *(const h8*)(r0 + 16 + koff));
      v16h a1 = mk16(*(const h8*)(r1 + koff), *(const h8*)(r1 + 16 + koff));
      acc0 = __builtin_amdgcn_wmma_f32_16x16x32_f16(false, a0, false, bfrag,
                                                    (short)0, acc0, false, false);
      acc1 = __builtin_amdgcn_wmma_f32_16x16x32_f16(false, a1, false, bfrag,
                                                    (short)0, acc1, false, false);
    }
  }
#endif

  // epilogue: C layout -> VGPR v holds (m = tile*16 + v + 8*hiHalf, n = lane&15)
  const int nloc = n0 + wn * 16 + rsel;
  const float bval = bias ? bias[nloc] : 0.0f;
#pragma unroll
  for (int s = 0; s < 2; ++s) {
    v8f acc = s ? acc1 : acc0;
    const int mtile = (wmBase + s) * 16;
#pragma unroll
    for (int v = 0; v < 8; ++v) {
      const int m = m0 + mtile + v + (hiHalf ? 8 : 0);
      float val = acc[v] + bval;
      if (relu) val = fmaxf(val, 0.0f);
      const size_t idx = (size_t)m * N + nloc;
      if (Cf) Cf[idx] = val;
      if (Ch) Ch[idx] = (half_t)val;
    }
  }
}

// ---------------------------------------------------------------------------
// Flash attention: softmax(Q K^T / sqrt(dh) [+mask]) V, per (b, h).
// Grid: (L/64, H, B); block 128 (4 waves); wave owns a 16-row Q tile.
// Key blocks of 32 staged in LDS (K row-major, V transposed), online softmax
// with group-of-16 shuffles (wave32), P round-trips through LDS to reach the
// WMMA A-fragment layout.
// ---------------------------------------------------------------------------
__launch_bounds__(128)
__global__ void attn_wmma_kernel(const half_t* __restrict__ Q,
                                 const half_t* __restrict__ Km,
                                 const half_t* __restrict__ Vm,
                                 const uint8_t* __restrict__ mask,
                                 half_t* __restrict__ ctx) {
  __shared__ __align__(16) half_t Ks[32 * 64];       // [key][dh]
  __shared__ __align__(16) half_t Vt[64 * 32];       // [dh][key]  (transposed)
  __shared__ __align__(16) half_t Ps[4][16 * 32];    // per-wave P tile

  const int tid    = threadIdx.x;
  const int lane   = tid & 31;
  const int wave   = tid >> 5;                       // 0..3
  const int b      = blockIdx.z;
  const int h      = blockIdx.y;
  const int q0     = blockIdx.x * 64 + wave * 16;
  const int rsel   = lane & 15;
  const int hiHalf = (lane >= 16);
  const float scl  = 0.125f;                         // 1/sqrt(64)
  const float NEGF = -2.14748352e9f;                 // 1 - 2^31

  // Q fragments: 16x64 as two 16x32 A-frags (loaded once)
  v16h qa[2];
  {
    const half_t* qrow = Q + ((size_t)(b * L_ + q0 + rsel)) * D_ + h * DH_;
    const int off = hiHalf ? 8 : 0;
#pragma unroll
    for (int kb = 0; kb < 2; ++kb) {
      h8 lo = *(const h8*)(qrow + kb * 32 + off);
      h8 hi = *(const h8*)(qrow + kb * 32 + 16 + off);
      qa[kb] = mk16(lo, hi);
    }
  }

  float rmax[8], rsum[8];
  v8f accO[4];
#pragma unroll
  for (int v = 0; v < 8; ++v) { rmax[v] = -3.0e38f; rsum[v] = 0.0f; }
#pragma unroll
  for (int nb = 0; nb < 4; ++nb) accO[nb] = vzero8();

  const size_t mbase0 = (size_t)(h * B_ + b) * L_ * L_;
  const int lr = tid >> 2;            // 0..31  (key row within block)
  const int lc = (tid & 3) * 16;      // 0,16,32,48 (dh column)

  for (int j0 = 0; j0 < L_; j0 += 32) {
    // ---- stage K (row major) and V (transposed) blocks in LDS ----
    {
      const size_t rowoff = ((size_t)(b * L_ + j0 + lr)) * D_ + h * DH_ + lc;
      h8 k0v = *(const h8*)(Km + rowoff);
      h8 k1v = *(const h8*)(Km + rowoff + 8);
      h8 v0v = *(const h8*)(Vm + rowoff);
      h8 v1v = *(const h8*)(Vm + rowoff + 8);
      __syncthreads();
      *(h8*)(&Ks[lr * 64 + lc])     = k0v;
      *(h8*)(&Ks[lr * 64 + lc + 8]) = k1v;
#pragma unroll
      for (int i = 0; i < 8; ++i) {
        Vt[(lc + i) * 32 + lr]     = v0v[i];
        Vt[(lc + 8 + i) * 32 + lr] = v1v[i];
      }
      __syncthreads();
    }

    // ---- S = Q * K^T (16x32, as two 16x16 accumulators) ----
    v8f S0 = vzero8(), S1 = vzero8();
#pragma unroll
    for (int kb = 0; kb < 2; ++kb) {
      const half_t* kp0 = &Ks[(rsel) * 64      + kb * 32 + (hiHalf ? 8 : 0)];
      const half_t* kp1 = &Ks[(16 + rsel) * 64 + kb * 32 + (hiHalf ? 8 : 0)];
      v16h bf0 = mk16(*(const h8*)kp0, *(const h8*)(kp0 + 16));
      v16h bf1 = mk16(*(const h8*)kp1, *(const h8*)(kp1 + 16));
      S0 = __builtin_amdgcn_wmma_f32_16x16x32_f16(false, qa[kb], false, bf0,
                                                  (short)0, S0, false, false);
      S1 = __builtin_amdgcn_wmma_f32_16x16x32_f16(false, qa[kb], false, bf1,
                                                  (short)0, S1, false, false);
    }

    // ---- scale + mask + online softmax; emit P tile to LDS ----
#pragma unroll
    for (int v = 0; v < 8; ++v) {
      const int m = v + (hiHalf ? 8 : 0);
      const size_t mb = mbase0 + (size_t)(q0 + m) * L_ + j0;
      float s0 = S0[v] * scl;
      float s1 = S1[v] * scl;
      if (!mask[mb + rsel])      s0 = NEGF;
      if (!mask[mb + 16 + rsel]) s1 = NEGF;
      float t = fmaxf(s0, s1);
#pragma unroll
      for (int off = 8; off >= 1; off >>= 1)
        t = fmaxf(t, __shfl_xor(t, off, 16));
      const float newmax = fmaxf(rmax[v], t);
      const float corr = __expf(rmax[v] - newmax);
      const float p0 = __expf(s0 - newmax);
      const float p1 = __expf(s1 - newmax);
      float psum = p0 + p1;
#pragma unroll
      for (int off = 8; off >= 1; off >>= 1)
        psum += __shfl_xor(psum, off, 16);
      rsum[v] = rsum[v] * corr + psum;
      rmax[v] = newmax;
#pragma unroll
      for (int nb = 0; nb < 4; ++nb) accO[nb][v] *= corr;
      Ps[wave][m * 32 + rsel]      = (half_t)p0;
      Ps[wave][m * 32 + 16 + rsel] = (half_t)p1;
    }
    __syncthreads();   // P visible across lanes before fragment re-load

    // ---- accO += P(16x32) @ V(32x64) ----
    v16h pf;
    {
      const half_t* pp = &Ps[wave][rsel * 32];
      const int off = hiHalf ? 8 : 0;
      pf = mk16(*(const h8*)(pp + off), *(const h8*)(pp + 16 + off));
    }
#pragma unroll
    for (int nb = 0; nb < 4; ++nb) {
      const half_t* vp = &Vt[(nb * 16 + rsel) * 32 + (hiHalf ? 8 : 0)];
      v16h bf = mk16(*(const h8*)vp, *(const h8*)(vp + 16));
      accO[nb] = __builtin_amdgcn_wmma_f32_16x16x32_f16(false, pf, false, bf,
                                                        (short)0, accO[nb], false, false);
    }
  }

  // ---- normalize and write ctx (f16) ----
  float invs[8];
#pragma unroll
  for (int v = 0; v < 8; ++v) invs[v] = 1.0f / rsum[v];
#pragma unroll
  for (int nb = 0; nb < 4; ++nb) {
#pragma unroll
    for (int v = 0; v < 8; ++v) {
      const int m = v + (hiHalf ? 8 : 0);
      const int n = nb * 16 + rsel;
      ctx[((size_t)(b * L_ + q0 + m)) * D_ + h * DH_ + n] =
          (half_t)(accO[nb][v] * invs[v]);
    }
  }
}

// ---------------------------------------------------------------------------
// Fused residual + LayerNorm over D=1024.  out = LN(xa + xb)*scale + bias.
// One block (256 threads) per row; optional f16 shadow copy.
// ---------------------------------------------------------------------------
__launch_bounds__(256)
__global__ void ln_kernel(const float* __restrict__ xa,
                          const float* __restrict__ xb,
                          const float* __restrict__ scale,
                          const float* __restrict__ bias,
                          float* __restrict__ outf,
                          half_t* __restrict__ outh) {
  __shared__ float s1[256];
  __shared__ float s2[256];
  const int row = blockIdx.x;
  const int tid = threadIdx.x;
  const size_t base = (size_t)row * D_;
  const int c = tid * 4;

  float4 va = *(const float4*)(xa + base + c);
  float4 vb = *(const float4*)(xb + base + c);
  const float v0 = va.x + vb.x, v1 = va.y + vb.y;
  const float v2 = va.z + vb.z, v3 = va.w + vb.w;

  s1[tid] = v0 + v1 + v2 + v3;
  s2[tid] = v0 * v0 + v1 * v1 + v2 * v2 + v3 * v3;
  __syncthreads();
  for (int off = 128; off > 0; off >>= 1) {
    if (tid < off) { s1[tid] += s1[tid + off]; s2[tid] += s2[tid + off]; }
    __syncthreads();
  }
  const float mean = s1[0] * (1.0f / D_);
  const float var  = s2[0] * (1.0f / D_) - mean * mean;
  const float rstd = rsqrtf(var + 1e-8f);

  const float o0 = (v0 - mean) * rstd * scale[c + 0] + bias[c + 0];
  const float o1 = (v1 - mean) * rstd * scale[c + 1] + bias[c + 1];
  const float o2 = (v2 - mean) * rstd * scale[c + 2] + bias[c + 2];
  const float o3 = (v3 - mean) * rstd * scale[c + 3] + bias[c + 3];
  *(float4*)(outf + base + c) = make_float4(o0, o1, o2, o3);
  if (outh) {
    outh[base + c + 0] = (half_t)o0;
    outh[base + c + 1] = (half_t)o1;
    outh[base + c + 2] = (half_t)o2;
    outh[base + c + 3] = (half_t)o3;
  }
}

// ---------------------------------------------------------------------------
// Host-side orchestration
// ---------------------------------------------------------------------------
static inline void launch_cast(const float* s, half_t* d, int n, hipStream_t st) {
  cast_f16_kernel<<<(n + 255) / 256, 256, 0, st>>>(s, d, n);
}

static inline void launch_cast_tr(const float* s, half_t* d, int K, int N,
                                  hipStream_t st) {
  dim3 g(N / 32, K / 32);
  cast_transpose_f16_kernel<<<g, 256, 0, st>>>(s, d, K, N);
}

extern "C" void kernel_launch(void* const* d_in, const int* in_sizes, int n_in,
                              void* d_out, int out_size, void* d_ws, size_t ws_size,
                              hipStream_t stream) {
  (void)in_sizes; (void)n_in; (void)out_size; (void)ws_size;

  const float*   x      = (const float*)d_in[0];
  const uint8_t* mask   = (const uint8_t*)d_in[1];
  const float*   Wq     = (const float*)d_in[2];
  const float*   Wk     = (const float*)d_in[3];
  const float*   Wv     = (const float*)d_in[4];
  const float*   Wo     = (const float*)d_in[5];
  const float*   W1     = (const float*)d_in[6];
  const float*   b1     = (const float*)d_in[7];
  const float*   W2     = (const float*)d_in[8];
  const float*   b2     = (const float*)d_in[9];
  const float*   scale1 = (const float*)d_in[10];
  const float*   bias1  = (const float*)d_in[11];
  const float*   scale2 = (const float*)d_in[12];
  const float*   bias2  = (const float*)d_in[13];

  const size_t ND  = (size_t)NT_ * D_;    // 4,194,304
  const size_t DD  = (size_t)D_ * D_;
  const size_t DF  = (size_t)D_ * DFF_;

  // workspace carve (f16 halves then f32); aliases exploit liveness.
  // All weight copies are TRANSPOSED to [N][K] so every GEMM B-tile is a
  // row-major rectangle (TDM/contiguous friendly).
  half_t* xh   = (half_t*)d_ws;
  half_t* WqT  = xh  + ND;
  half_t* WkT  = WqT + DD;
  half_t* WvT  = WkT + DD;
  half_t* WoT  = WvT + DD;
  half_t* W1T  = WoT + DD;                 // [DFF][D]
  half_t* W2T  = W1T + DF;                 // [D][DFF]
  half_t* Qh   = W2T + DF;
  half_t* Kh   = Qh  + ND;
  half_t* Vh   = Kh  + ND;
  half_t* ctxh = Vh  + ND;
  half_t* ff1h = Qh;                       // alias: Q/K/V/ctx dead before FFN
  float*  attnf = (float*)(ctxh + ND);
  float*  ff2f  = attnf;                   // alias: attnf dead before FFN-2
  float*  a1f   = attnf + ND;
  half_t* a1h   = (half_t*)(a1f + ND);
  float*  outp  = (float*)d_out;

  // 1) casts: activations plain, weights cast+transpose
  launch_cast(x, xh, (int)ND, stream);
  launch_cast_tr(Wq, WqT, D_, D_, stream);
  launch_cast_tr(Wk, WkT, D_, D_, stream);
  launch_cast_tr(Wv, WvT, D_, D_, stream);
  launch_cast_tr(Wo, WoT, D_, D_, stream);
  launch_cast_tr(W1, W1T, D_, DFF_, stream);   // [D][DFF] -> [DFF][D]
  launch_cast_tr(W2, W2T, DFF_, D_, stream);   // [DFF][D] -> [D][DFF]

  // 2) Q/K/V projections (f16 out)
  {
    dim3 g(D_ / 64, NT_ / 64);
    gemm_f16_wmma<<<g, 256, 0, stream>>>(xh, WqT, nullptr, nullptr, Qh, NT_, D_, D_, 0);
    gemm_f16_wmma<<<g, 256, 0, stream>>>(xh, WkT, nullptr, nullptr, Kh, NT_, D_, D_, 0);
    gemm_f16_wmma<<<g, 256, 0, stream>>>(xh, WvT, nullptr, nullptr, Vh, NT_, D_, D_, 0);
  }

  // 3) attention
  {
    dim3 g(L_ / 64, H_, B_);
    attn_wmma_kernel<<<g, 128, 0, stream>>>(Qh, Kh, Vh, mask, ctxh);
  }

  // 4) output projection (f32 out)
  {
    dim3 g(D_ / 64, NT_ / 64);
    gemm_f16_wmma<<<g, 256, 0, stream>>>(ctxh, WoT, nullptr, attnf, nullptr, NT_, D_, D_, 0);
  }

  // 5) a1 = LN(x + selfattn)
  ln_kernel<<<NT_, 256, 0, stream>>>(x, attnf, scale1, bias1, a1f, a1h);

  // 6) ff1 = relu(a1 @ W1 + b1)  (f16 out)
  {
    dim3 g(DFF_ / 64, NT_ / 64);
    gemm_f16_wmma<<<g, 256, 0, stream>>>(a1h, W1T, b1, nullptr, ff1h, NT_, DFF_, D_, 1);
  }

  // 7) ff2 = ff1 @ W2 + b2  (f32 out)
  {
    dim3 g(D_ / 64, NT_ / 64);
    gemm_f16_wmma<<<g, 256, 0, stream>>>(ff1h, W2T, b2, ff2f, nullptr, NT_, D_, DFF_, 0);
  }

  // 8) out = LN(a1 + ff2)
  ln_kernel<<<NT_, 256, 0, stream>>>(a1f, ff2f, scale2, bias2, outp, nullptr);
}
